// LSTM_42477226558055
// MI455X (gfx1250) — matrix operand order repaired and checked by reading, not compiled, not run on hardware
//
#include <hip/hip_runtime.h>

// LSTM forward scan for MI455X (gfx1250, wave32, WMMA, TDM-staged weights).
//
// Shapes: B=64, T=256, D_IN=512, H=1024, K = D_IN+H = 1536, gates = 4H.
// Per step: [64x1536] @ [1536x4096] (bf16 WMMA, f32 accum) fused with gate
// nonlinearities and the c/h state update; 256 stream-ordered step kernels.
//
// Step-kernel mapping: 64 blocks (16 h-cols each) x 256 threads (8 waves,
// 2 waves/SIMD). Wave = (m-tile 0..3, gate-pair 0..1): pair 0 accumulates
// {f,i}, pair 1 {c~,o} for the same 16x16 tile. The block's 192KB weight
// slice (pre-shuffled to the exact LDS fragment layout by the prep kernel)
// is DMA'd into LDS by one TENSOR_LOAD_TO_LDS (TDM) issued by wave 0
// (fallback: coalesced manual copy), then the fully-unrolled K loop reads
// B fragments via conflict-free ds_load_b128 with register double-buffering
// of both A (global) and B (LDS) so loads pipeline ahead of the WMMAs.
// Epilogue: gate tiles -> 16KB LDS gate buffer -> barrier -> cooperative
// elementwise sigmoid/tanh/c/h update.

typedef __attribute__((ext_vector_type(16))) __bf16 v16bf;
typedef __attribute__((ext_vector_type(8)))  float  v8f;
typedef __attribute__((ext_vector_type(4)))  unsigned int u32x4;
typedef __attribute__((ext_vector_type(8)))  int i32x8;
typedef __attribute__((ext_vector_type(4)))  int i32x4;

#define B_SZ 64
#define T_SZ 256
#define DIN  512
#define H_SZ 1024
#define KTOT (DIN + H_SZ)      /* 1536 */
#define NKB  (KTOT / 32)       /* 48 K-iterations */
#define WSLICE_HALFS (NKB * 4 * 2 * 32 * 8)   /* 98304 halfs = 196608 B per block */
#define LDS_W_BYTES  (WSLICE_HALFS * 2)       /* 196608 */
#define LDS_G_BYTES  (4 * B_SZ * 16 * 4)      /* 16384: gbuf[4][64][16] f32 */
#define LDS_TOTAL    (LDS_W_BYTES + LDS_G_BYTES)
#define WSLICE_D64   (LDS_W_BYTES / 8)        /* 24576 8-byte elements */

#if defined(__has_builtin)
#if __has_builtin(__builtin_amdgcn_tensor_load_to_lds) && \
    __has_builtin(__builtin_amdgcn_s_wait_tensorcnt)
#define USE_TDM 1
#endif
#endif

union FragBF {
  v16bf v;
  uint4 q[2]; // 32 bytes = 16 bf16 per lane
};

__device__ __forceinline__ float fast_sigmoid(float x) {
  return 1.0f / (1.0f + __expf(-x));
}

__device__ __forceinline__ v8f wmma_bf16(const FragBF& a, const FragBF& b, v8f c) {
  return __builtin_amdgcn_wmma_f32_16x16x32_bf16(false, a.v, false, b.v,
                                                 (short)0, c, false, false);
}

__device__ __forceinline__ void load_a(const __bf16* p, int abase, FragBF& a) {
  a.q[0] = *(const uint4*)(p + abase);
  a.q[1] = *(const uint4*)(p + abase + 16);
}

__device__ __forceinline__ void load_b(const uint4* sb, int idx,
                                       FragBF& b0, FragBF& b1) {
  b0.q[0] = sb[idx];
  b0.q[1] = sb[idx + 32];
  b1.q[0] = sb[idx + 64];
  b1.q[1] = sb[idx + 96];
}

// ---- prep: x [b][t][k] f32 -> xb [t][b][k] bf16 -----------------------------
__global__ void lstm_prep_x(const float* __restrict__ x, __bf16* __restrict__ xb) {
  int idx = blockIdx.x * blockDim.x + threadIdx.x;
  const int total = B_SZ * T_SZ * DIN;
  if (idx >= total) return;
  int k = idx % DIN;
  int t = (idx / DIN) % T_SZ;
  int b = idx / (DIN * T_SZ);
  xb[((size_t)t * B_SZ + b) * DIN + k] = (__bf16)x[idx];
}

// ---- prep: W_g [k][n] f32 -> Wt in LDS-fragment-shuffled bf16 layout --------
// flat out index = ((((nblk*NKB + kb)*4 + g)*2 + j2)*32 + lane)*8 + j8
// maps to W_g[k][n] with n = nblk*16 + (lane&15),
//                        k = kb*32 + j2*16 + (lane>>4)*8 + j8.
__global__ void lstm_prep_w(const float* __restrict__ Wf, const float* __restrict__ Wi,
                            const float* __restrict__ Wc, const float* __restrict__ Wo,
                            __bf16* __restrict__ Wt) {
  int idx = blockIdx.x * blockDim.x + threadIdx.x;
  const int total = 64 * WSLICE_HALFS;       // 4*H*KTOT = 6.29M
  if (idx >= total) return;
  int nblk = idx / WSLICE_HALFS;
  int r    = idx % WSLICE_HALFS;
  int kb   = r / 2048;            // 4*2*32*8
  int r2   = r % 2048;
  int g    = r2 / 512;
  int r3   = r2 % 512;
  int j2   = r3 / 256;
  int r4   = r3 % 256;
  int lane = r4 / 8;
  int j8   = r4 % 8;
  int n = nblk * 16 + (lane & 15);
  int k = kb * 32 + j2 * 16 + (lane >> 4) * 8 + j8;
  const float* W = (g == 0) ? Wf : (g == 1) ? Wi : (g == 2) ? Wc : Wo;
  Wt[(size_t)idx] = (__bf16)W[(size_t)k * H_SZ + n];
}

// ---- prep: zero h0 (bf16) and c0 (f32) --------------------------------------
__global__ void lstm_init_state(__bf16* __restrict__ h0, float* __restrict__ c0) {
  int idx = blockIdx.x * blockDim.x + threadIdx.x;
  if (idx < B_SZ * H_SZ) {
    h0[idx] = (__bf16)0.0f;
    c0[idx] = 0.0f;
  }
}

// ---- one LSTM timestep ------------------------------------------------------
__global__ void __launch_bounds__(256)
lstm_step(const __bf16* __restrict__ xb,      // [T][B][DIN] bf16
          const __bf16* __restrict__ Wt,      // shuffled weight layout (see prep)
          const __bf16* __restrict__ h_prev,  // [B][H] bf16
          __bf16* __restrict__ h_next,        // [B][H] bf16
          float* __restrict__ c_state,        // [B][H] f32 (in-place)
          const float* __restrict__ b_f, const float* __restrict__ b_i,
          const float* __restrict__ b_c, const float* __restrict__ b_o,
          float* __restrict__ out,            // [B][H] f32
          int t) {
  extern __shared__ __align__(16) char smem[];   // 196608B weights + 16384B gates

  const int tid   = threadIdx.x;
  const int lane  = tid & 31;
  const int wave  = tid >> 5;
  const int mt    = wave >> 1;          // m-tile 0..3
  const int gh    = wave & 1;           // 0 -> gates {f,i}, 1 -> {c~,o}
  const int m0    = mt * 16;
  const int lhalf = lane & 15;
  const int lsel  = lane >> 4;

  const __bf16* wslice = Wt + (size_t)blockIdx.x * WSLICE_HALFS;

  // ---- stage this block's 192KB weight slice into LDS ----
#if USE_TDM
  if (tid < 32) {
    // D# per CDNA5 ISA ch.8: 1-row 2D tensor, data_size=8B,
    // tensor_dim0 = tile_dim0 = 24576 (8B units), contiguous.
    const unsigned long long ga = (unsigned long long)(uintptr_t)wslice;
    const unsigned int lds_base = (unsigned int)(uintptr_t)smem; // LDS byte offset
    u32x4 g0;
    g0[0] = 1u;                                  // count=1 (valid descriptor)
    g0[1] = lds_base;                            // lds_addr
    g0[2] = (unsigned int)(ga & 0xFFFFFFFFu);    // global_addr[31:0]
    g0[3] = (unsigned int)((ga >> 32) & 0x01FFFFFFu) | 0x80000000u; // [56:32]|type=2
    i32x8 g1;
    g1[0] = 0x00030000;            // data_size=3 (8B); mask/pad/iterate=0
    g1[1] = (int)(WSLICE_D64 << 16);   // tensor_dim0[15:0] << 16 (bits 63:48)
    g1[2] = 0x00010000;            // tensor_dim0[31:16]=0 | tensor_dim1=1
    g1[3] = (int)(WSLICE_D64 << 16);   // tile_dim0 (bits 127:112)
    g1[4] = 1;                     // tile_dim1 = 1
    g1[5] = WSLICE_D64;            // tensor_dim0_stride[31:0]
    g1[6] = (int)(WSLICE_D64 << 16);   // stride hi=0 | tensor_dim1_stride lo16
    g1[7] = 0;
    i32x4 z4 = {0, 0, 0, 0};
#if __clang_major__ >= 23
    i32x8 z8 = {0, 0, 0, 0, 0, 0, 0, 0};
    __builtin_amdgcn_tensor_load_to_lds(g0, g1, z4, z4, z8, 0);
#else
    __builtin_amdgcn_tensor_load_to_lds(g0, g1, z4, z4, 0);
#endif
    __builtin_amdgcn_s_wait_tensorcnt((short)0);
  }
#else
  {
    const uint4* src = (const uint4*)wslice;
    uint4* dst = (uint4*)smem;
#pragma unroll 4
    for (int i = tid; i < WSLICE_HALFS / 8; i += 256) dst[i] = src[i];
  }
#endif
  __syncthreads();

  const uint4* sb = (const uint4*)smem;
  float* gbuf = (float*)(smem + LDS_W_BYTES);    // [4][64][16] f32

  v8f acc0 = {}, acc1 = {};

  // A source rows (16-bit A 16x32 layout: lane -> row m0+(l&15),
  // chunks at k0 + (l>>4)*8 and +16).
  const __bf16* xrow = xb + ((size_t)t * B_SZ + (m0 + lhalf)) * DIN;
  const __bf16* hrow = h_prev + (size_t)(m0 + lhalf) * H_SZ;
  const int abase = lsel * 8;

  // B fragment uint4 index: kb*256 + g*64 + j2*32 + lane.
  const int bofs = gh * 128 + lane;

  // ---- K loop: fully unrolled so the x/h source switch, the clamped
  // prefetch index and the buffer rotation all resolve at compile time
  // (no cndmask chains, no v_mov rotation, immediate ds/global offsets). ----
  FragBF a_cur, a_nxt, b0c, b0n, b1c, b1n;
  load_a(xrow, abase, a_cur);
  load_b(sb, bofs, b0c, b1c);
#pragma unroll
  for (int i = 0; i < NKB; ++i) {
    const int j = (i + 1 < NKB) ? (i + 1) : i;     // clamp (folds per-iteration)
    const __bf16* an = (j < DIN / 32) ? (xrow + j * 32)
                                      : (hrow + (j - DIN / 32) * 32);
    load_a(an, abase, a_nxt);
    load_b(sb, j * 256 + bofs, b0n, b1n);
    acc0 = wmma_bf16(a_cur, b0c, acc0);
    acc1 = wmma_bf16(a_cur, b1c, acc1);
    a_cur = a_nxt;
    b0c = b0n;
    b1c = b1n;
  }

  // ---- dump gate tiles to LDS gate buffer ----
  // C/D layout: (vgpr r, lane l) -> row m0 + r + (l>>4)*8, col (l&15).
  const int g0i = gh * 2;
#pragma unroll
  for (int r = 0; r < 8; ++r) {
    const int m = m0 + r + lsel * 8;
    gbuf[((g0i + 0) * B_SZ + m) * 16 + lhalf] = acc0[r];
    gbuf[((g0i + 1) * B_SZ + m) * 16 + lhalf] = acc1[r];
  }
  __syncthreads();

  // ---- cooperative elementwise LSTM update over the 64x16 output tile ----
#pragma unroll
  for (int e = tid; e < B_SZ * 16; e += 256) {
    const int m  = e >> 4;
    const int nl = e & 15;
    const int n  = blockIdx.x * 16 + nl;
    const float fg = fast_sigmoid(gbuf[(0 * B_SZ + m) * 16 + nl] + b_f[n]);
    const float ig = fast_sigmoid(gbuf[(1 * B_SZ + m) * 16 + nl] + b_i[n]);
    const float cg = tanhf(gbuf[(2 * B_SZ + m) * 16 + nl] + b_c[n]);
    const float og = fast_sigmoid(gbuf[(3 * B_SZ + m) * 16 + nl] + b_o[n]);
    const size_t off = (size_t)m * H_SZ + n;
    const float cn_ = fg * c_state[off] + ig * cg;
    c_state[off] = cn_;
    const float h = og * tanhf(cn_);
    h_next[off] = (__bf16)h;
    if (t == T_SZ - 1) out[off] = h;
  }
}

extern "C" void kernel_launch(void* const* d_in, const int* in_sizes, int n_in,
                              void* d_out, int out_size, void* d_ws, size_t ws_size,
                              hipStream_t stream) {
  const float* x  = (const float*)d_in[0];
  const float* Wf = (const float*)d_in[1];
  const float* bf = (const float*)d_in[2];
  const float* Wi = (const float*)d_in[3];
  const float* bi = (const float*)d_in[4];
  const float* Wc = (const float*)d_in[5];
  const float* bc = (const float*)d_in[6];
  const float* Wo = (const float*)d_in[7];
  const float* bo = (const float*)d_in[8];
  float* out = (float*)d_out;

  // Workspace carve-up (~29.9 MB total).
  char* ws = (char*)d_ws;
  size_t off = 0;
  __bf16* xb = (__bf16*)(ws + off); off += (size_t)T_SZ * B_SZ * DIN * 2;   // 16 MB
  __bf16* Wt = (__bf16*)(ws + off); off += (size_t)64 * WSLICE_HALFS * 2;   // 12.6 MB
  __bf16* hA = (__bf16*)(ws + off); off += (size_t)B_SZ * H_SZ * 2;         // 128 KB
  __bf16* hB = (__bf16*)(ws + off); off += (size_t)B_SZ * H_SZ * 2;         // 128 KB
  float*  cs = (float*)(ws + off);  off += (size_t)B_SZ * H_SZ * 4;         // 256 KB

  lstm_prep_x<<<(B_SZ * T_SZ * DIN + 255) / 256, 256, 0, stream>>>(x, xb);
  lstm_prep_w<<<(64 * WSLICE_HALFS + 255) / 256, 256, 0, stream>>>(Wf, Wi, Wc, Wo, Wt);
  lstm_init_state<<<(B_SZ * H_SZ + 255) / 256, 256, 0, stream>>>(hA, cs);

  for (int t = 0; t < T_SZ; ++t) {
    const __bf16* hp = (t & 1) ? hB : hA;
    __bf16*       hn = (t & 1) ? hA : hB;
    lstm_step<<<H_SZ / 16, 256, LDS_TOTAL, stream>>>(xb, Wt, hp, hn, cs,
                                                     bf, bi, bc, bo, out, t);
  }
}
